// NeighborAttention_6339371729564
// MI455X (gfx1250) — matrix operand order repaired and checked by reading, not compiled, not run on hardware
//
#include <hip/hip_runtime.h>
#include <hip/hip_bf16.h>

#define NEG_INF (-3.402823466e+38f)

typedef __attribute__((ext_vector_type(8)))  __bf16 v8bf;
typedef __attribute__((ext_vector_type(16))) __bf16 v16bf;
typedef __attribute__((ext_vector_type(8)))  float  v8f;
typedef __attribute__((ext_vector_type(4)))  float  v4f;
typedef __attribute__((ext_vector_type(4)))  unsigned int tdm_v4u;
typedef __attribute__((ext_vector_type(8)))  int          tdm_v8i;
typedef __attribute__((ext_vector_type(4)))  int          tdm_v4i;

// ---- constants for this problem instance ----
#define BATCH 4
#define NNODE 4096
#define KNBR  32
#define CIN   128
#define COUT  128
#define HEADS 4
#define DHEAD 32
#define BN    (BATCH * NNODE)   // 16384
#define NT    2                 // nodes per workgroup in fused kernel

#if __has_builtin(__builtin_amdgcn_tensor_load_to_lds) && __has_builtin(__builtin_amdgcn_s_wait_tensorcnt)
#define HAVE_TDM 1
#else
#define HAVE_TDM 0
#endif

// Build A-tile register image (16x32 bf16) from two contiguous 8-half chunks.
// ISA layout: lane<16 -> row=lane%16, K = {kA..kA+7, kA+16..kA+23}, kA = 8*(lane>=16).
__device__ __forceinline__ v16bf make_a16x32(const __bf16* lo_p) {
    v8bf lo = *(const v8bf*)(lo_p);
    v8bf hi = *(const v8bf*)(lo_p + 16);
    return __builtin_shufflevector(lo, hi, 0,1,2,3,4,5,6,7,8,9,10,11,12,13,14,15);
}

// Convert 16 fp32 (groups [0..7] and [16..23] relative to p) into an A-tile image.
__device__ __forceinline__ v16bf cvt_a16x32_f32(const float* p) {
    v16bf r;
#pragma unroll
    for (int i = 0; i < 8; ++i) {
        r[i]     = (__bf16)p[i];
        r[i + 8] = (__bf16)p[i + 16];
    }
    return r;
}

__device__ __forceinline__ v8f wmma_bf16(v16bf a, v16bf b, v8f c) {
    return __builtin_amdgcn_wmma_f32_16x16x32_bf16(false, a, false, b, (short)0, c, false, false);
}

// ---------------------------------------------------------------------------
// K0: convert weights fp32 -> bf16 into workspace (stay resident in L2)
// ---------------------------------------------------------------------------
__global__ void k_cvt_weights(const float* __restrict__ WQ, const float* __restrict__ WK,
                              const float* __restrict__ WV, const float* __restrict__ WO,
                              __bf16* __restrict__ WQb, __bf16* __restrict__ WKb,
                              __bf16* __restrict__ WVb, __bf16* __restrict__ WOb) {
    int i = blockIdx.x * 256 + threadIdx.x;
    if (i < COUT * COUT) {
        WQb[i] = (__bf16)WQ[i];
        WKb[i] = (__bf16)WK[i];
        WVb[i] = (__bf16)WV[i];
    }
    if (i < COUT * 3 * COUT) WOb[i] = (__bf16)WO[i];
}

// ---------------------------------------------------------------------------
// K1: hq = X[BN,128] @ WQ^T -> fp32 workspace.  WG = 64 rows x 128 cols.
// ---------------------------------------------------------------------------
__global__ void k_qproj(const float* __restrict__ X, const __bf16* __restrict__ WQb,
                        float* __restrict__ hq) {
    const int t = threadIdx.x, w = t >> 5, lane = t & 31;
    const int l16 = lane & 15, hiL = lane >> 4;
    const int mt = w & 3, ng = w >> 2;
    const int mrow = blockIdx.x * 64 + mt * 16 + l16;
    const int kA = hiL ? 8 : 0, kB = hiL ? 8 : 0;

    v8f acc[4] = {};
#pragma unroll
    for (int k0 = 0; k0 < CIN; k0 += 32) {
        v16bf a = cvt_a16x32_f32(X + (size_t)mrow * CIN + k0 + kA);
#pragma unroll
        for (int nt = 0; nt < 4; ++nt) {
            int o = ng * 64 + nt * 16 + l16;
            v16bf b = make_a16x32(WQb + (size_t)o * CIN + k0 + kB);
            acc[nt] = wmma_bf16(a, b, acc[nt]);
        }
    }
#pragma unroll
    for (int nt = 0; nt < 4; ++nt) {
        int o = ng * 64 + nt * 16 + l16;
#pragma unroll
        for (int r = 0; r < 8; ++r) {
            int M = blockIdx.x * 64 + mt * 16 + hiL * 8 + r;
            hq[(size_t)M * COUT + o] = acc[nt][r];
        }
    }
}

// ---------------------------------------------------------------------------
// K2: fused kernel, NT=2 nodes per WG (256 thr, 8 waves).
//   phase0: TDM-load E tile (64x128 fp32) -> LDS  (TENSORcnt-tracked DMA)
//   phase1: hk = E@WK^T (waves 0-3), hv = E@WV^T (waves 4-7) via WMMA bf16
//   phase2: masked softmax: one wave per (node,head), lane = neighbor
//   phase3: sum/mean/max aggregation -> cat[384] bf16 per node
// ---------------------------------------------------------------------------
__global__ void __launch_bounds__(256) k_node(const float* __restrict__ E,
                                              const int* __restrict__ maskp,
                                              const float* __restrict__ hq,
                                              const __bf16* __restrict__ WKb,
                                              const __bf16* __restrict__ WVb,
                                              __bf16* __restrict__ catb) {
    __shared__ __align__(16) float EfS[NT * KNBR * CIN];   // 32 KB fp32
    __shared__ __align__(16) float hkS[NT * KNBR * COUT];  // 32 KB
    __shared__ __align__(16) float hvS[NT * KNBR * COUT];  // 32 KB
    __shared__ float qvS[NT * COUT];
    __shared__ float attS[NT * HEADS * KNBR];
    __shared__ float asumS[NT * HEADS];

    const int t = threadIdx.x;
    const int bn0 = blockIdx.x * NT;
    const float* Eg = E + (size_t)bn0 * (KNBR * CIN);

    // ---- phase 0: stage E tile (NT*32 rows x 128 fp32) into LDS ----
#if HAVE_TDM
    if (t < 32) {
        // Tensor DMA descriptor (ISA cdna5 ch.8): 2D tile, fp32 elements.
        unsigned long long ga = (unsigned long long)(uintptr_t)Eg;
        unsigned int lds = (unsigned int)(uintptr_t)(&EfS[0]);
        tdm_v4u g0;
        g0[0] = 1u;                                  // count=1, user descriptor
        g0[1] = lds;                                 // lds_addr (bytes)
        g0[2] = (unsigned int)ga;                    // global_addr[31:0]
        g0[3] = (unsigned int)((ga >> 32) & 0x1FFFFFFu) | 0x80000000u;  // addr[56:32], type=2
        tdm_v8i g1;
        g1[0] = (int)(2u << 16);                     // workgroup_mask=0, data_size=4B
        g1[1] = (int)(128u << 16);                   // tensor_dim0 = 128 (lo16 in [31:16])
        g1[2] = (int)((NT * KNBR) << 16);            // dim0 hi=0 | tensor_dim1 = 64
        g1[3] = (int)(128u << 16);                   // dim1 hi=0 | tile_dim0 = 128
        g1[4] = NT * KNBR;                           // tile_dim1 = 64, tile_dim2 = 0
        g1[5] = 128;                                 // tensor_dim0_stride = 128 elements
        g1[6] = 0;
        g1[7] = 0;
        tdm_v4i gz = {0, 0, 0, 0};
#if defined(__clang_major__) && (__clang_major__ >= 23)
        tdm_v8i gz8 = {0, 0, 0, 0, 0, 0, 0, 0};
        __builtin_amdgcn_tensor_load_to_lds(g0, g1, gz, gz, gz8, 0);
#else
        __builtin_amdgcn_tensor_load_to_lds(g0, g1, gz, gz, 0);
#endif
        __builtin_amdgcn_s_wait_tensorcnt(0);        // DMA done before barrier publish
    }
#else
    // Fallback: manual vector staging.
#pragma unroll
    for (int j = 0; j < 8; ++j) {
        int idx = t * 32 + j * 4;
        *(v4f*)(EfS + idx) = *(const v4f*)(Eg + idx);
    }
#endif
    qvS[t] = hq[(size_t)bn0 * COUT + t];             // 256 = NT*128 contiguous
    __syncthreads();

    // ---- phase 1: WMMA projections hk / hv ----
    {
        const int w = t >> 5, lane = t & 31;
        const int l16 = lane & 15, hiL = lane >> 4;
        const int mat = w >> 2;          // 0 -> hk, 1 -> hv
        const int ntp = w & 3;           // n-tile pair
        const int kA = hiL ? 8 : 0, kB = hiL ? 8 : 0;
        const __bf16* Wb = mat ? WVb : WKb;

        v8f acc[4][2] = {};
#pragma unroll
        for (int k0 = 0; k0 < CIN; k0 += 32) {
            v16bf a[4], b[2];
#pragma unroll
            for (int mtl = 0; mtl < 4; ++mtl)
                a[mtl] = cvt_a16x32_f32(EfS + (mtl * 16 + l16) * CIN + k0 + kA);
#pragma unroll
            for (int nt = 0; nt < 2; ++nt) {
                int ncol = (ntp * 2 + nt) * 16 + l16;
                b[nt] = make_a16x32(Wb + (size_t)ncol * CIN + k0 + kB);
            }
#pragma unroll
            for (int mtl = 0; mtl < 4; ++mtl)
#pragma unroll
                for (int nt = 0; nt < 2; ++nt)
                    acc[mtl][nt] = wmma_bf16(a[mtl], b[nt], acc[mtl][nt]);
        }
        float* dst = mat ? hvS : hkS;
#pragma unroll
        for (int mtl = 0; mtl < 4; ++mtl)
#pragma unroll
            for (int nt = 0; nt < 2; ++nt) {
                int ncol = (ntp * 2 + nt) * 16 + l16;
#pragma unroll
                for (int r = 0; r < 8; ++r) {
                    int M = mtl * 16 + hiL * 8 + r;      // 0..63
                    dst[M * COUT + ncol] = acc[mtl][nt][r];
                }
            }
    }
    __syncthreads();

    // ---- phase 2: scores + masked softmax (wave = (node,head), lane = k) ----
    {
        const int nl = t >> 7;
        const int h  = (t >> 5) & 3;
        const int k  = t & 31;
        const float* hkrow = hkS + (nl * KNBR + k) * COUT + h * DHEAD;
        const float* qv    = qvS + nl * COUT + h * DHEAD;
        float s = 0.f;
#pragma unroll
        for (int dd = 0; dd < DHEAD; ++dd) s += qv[dd] * hkrow[dd];
        s *= 0.17677669529663687f;  // 1/sqrt(32)
        const int m = maskp[(size_t)(bn0 + nl) * KNBR + k];
        float logit = (m > 0) ? s : NEG_INF;
        float mx = logit;
#pragma unroll
        for (int off = 16; off; off >>= 1) mx = fmaxf(mx, __shfl_xor(mx, off, 32));
        float e = __expf(logit - mx);
        float se = e;
#pragma unroll
        for (int off = 16; off; off >>= 1) se += __shfl_xor(se, off, 32);
        float av = (m > 0) ? (e / se) : 0.f;
        attS[t] = av;
        float ss = av;
#pragma unroll
        for (int off = 16; off; off >>= 1) ss += __shfl_xor(ss, off, 32);
        if (k == 0) asumS[nl * HEADS + h] = ss;
    }
    __syncthreads();

    // ---- phase 3: aggregation over neighbors; emit cat = [mean|sum|max] ----
    {
        const int nl = t >> 7, c = t & 127, h = c >> 5;
        float aS = 0.f, aM = -INFINITY;
#pragma unroll
        for (int k = 0; k < KNBR; ++k) {
            float wv = attS[nl * 128 + h * KNBR + k] * hvS[(nl * KNBR + k) * COUT + c];
            aS += wv;
            aM = fmaxf(aM, wv);
        }
        float mean = aS / (asumS[nl * HEADS + h] + 1e-8f);
        __bf16* cb = catb + (size_t)(bn0 + nl) * (3 * COUT);
        cb[c]            = (__bf16)mean;
        cb[COUT + c]     = (__bf16)aS;
        cb[2 * COUT + c] = (__bf16)aM;
    }
}

// ---------------------------------------------------------------------------
// K3: out = cat[BN,384] @ WO^T -> fp32 d_out.  Same tiling as K1, Kdim=384.
// ---------------------------------------------------------------------------
__global__ void k_outproj(const __bf16* __restrict__ catb, const __bf16* __restrict__ WOb,
                          float* __restrict__ out) {
    const int t = threadIdx.x, w = t >> 5, lane = t & 31;
    const int l16 = lane & 15, hiL = lane >> 4;
    const int mt = w & 3, ng = w >> 2;
    const int mrow = blockIdx.x * 64 + mt * 16 + l16;
    const int kA = hiL ? 8 : 0, kB = hiL ? 8 : 0;
    const int KD = 3 * COUT;

    v8f acc[4] = {};
#pragma unroll
    for (int k0 = 0; k0 < 3 * COUT; k0 += 32) {
        v16bf a = make_a16x32(catb + (size_t)mrow * KD + k0 + kA);
#pragma unroll
        for (int nt = 0; nt < 4; ++nt) {
            int o = ng * 64 + nt * 16 + l16;
            v16bf b = make_a16x32(WOb + (size_t)o * KD + k0 + kB);
            acc[nt] = wmma_bf16(a, b, acc[nt]);
        }
    }
#pragma unroll
    for (int nt = 0; nt < 4; ++nt) {
        int o = ng * 64 + nt * 16 + l16;
#pragma unroll
        for (int r = 0; r < 8; ++r) {
            int M = blockIdx.x * 64 + mt * 16 + hiL * 8 + r;
            out[(size_t)M * COUT + o] = acc[nt][r];
        }
    }
}

// ---------------------------------------------------------------------------
extern "C" void kernel_launch(void* const* d_in, const int* in_sizes, int n_in,
                              void* d_out, int out_size, void* d_ws, size_t ws_size,
                              hipStream_t stream) {
    const float* hX   = (const float*)d_in[0];
    const float* hE   = (const float*)d_in[1];
    const int*   mask = (const int*)d_in[2];
    const float* WQ   = (const float*)d_in[3];
    const float* WK   = (const float*)d_in[4];
    const float* WV   = (const float*)d_in[5];
    const float* WO   = (const float*)d_in[6];
    float* out = (float*)d_out;

    char* ws = (char*)d_ws;
    __bf16* WQb = (__bf16*)(ws);                         // 32 KB
    __bf16* WKb = (__bf16*)(ws + 32768);                 // 32 KB
    __bf16* WVb = (__bf16*)(ws + 65536);                 // 32 KB
    __bf16* WOb = (__bf16*)(ws + 98304);                 // 96 KB
    float*  hq  = (float*)(ws + 196608);                 // 8 MB  (BN*128 f32)
    __bf16* cat = (__bf16*)(ws + 196608 + (size_t)BN * COUT * 4);  // 12.6 MB (BN*384 bf16)

    k_cvt_weights<<<(COUT * 3 * COUT + 255) / 256, 256, 0, stream>>>(WQ, WK, WV, WO,
                                                                     WQb, WKb, WVb, WOb);
    k_qproj<<<BN / 64, 256, 0, stream>>>(hX, WQb, hq);
    k_node<<<BN / NT, 256, 0, stream>>>(hE, mask, hq, WKb, WVb, cat);
    k_outproj<<<BN / 64, 256, 0, stream>>>(cat, WOb, out);
}
